// NeuralCDE_9225589752314
// MI455X (gfx1250) — matrix operand order repaired and checked by reading, not compiled, not run on hardware
//
#include <hip/hip_runtime.h>
#include <hip/hip_bf16.h>

// Problem constants
#define BB 256
#define TT 64
#define DD 16
#define SS 64
#define HH 128
#define OO 8
#define NSTEP 63   // T-1

typedef __attribute__((ext_vector_type(16))) __bf16 v16bf;
typedef __attribute__((ext_vector_type(8)))  float  v8f;

struct U32B { uint4 lo, hi; };   // 32 bytes == v16bf

__device__ __forceinline__ unsigned short f2bf(float x) {
  unsigned u = __builtin_bit_cast(unsigned, x);
  u += 0x7FFFu + ((u >> 16) & 1u);          // round-to-nearest-even
  return (unsigned short)(u >> 16);
}

// Fast softplus: max(x,0) + ln(1+exp(-|x|)).  1+e is in [1,2] so the raw
// v_exp_f32 / v_log_f32 (exp2/log2) trans ops need no denorm fixup.
__device__ __forceinline__ float softplus_f(float x) {
#if __has_builtin(__builtin_amdgcn_exp2f) && __has_builtin(__builtin_amdgcn_logf)
  float e = __builtin_amdgcn_exp2f(-1.4426950408889634f * fabsf(x));
  float l = __builtin_amdgcn_logf(1.0f + e);
  return fmaf(0.6931471805599453f, l, fmaxf(x, 0.0f));
#else
  return fmaxf(x, 0.0f) + __logf(1.0f + __expf(-fabsf(x)));
#endif
}

// Hardware tanh if the gfx1250 builtin exists; otherwise a short native-exp
// form: tanh(x) = sign(x) * (1 - 2e/(1+e)), e = exp(-2|x|).
__device__ __forceinline__ float fast_tanh(float x) {
#if __has_builtin(__builtin_amdgcn_tanhf)
  return __builtin_amdgcn_tanhf(x);
#elif __has_builtin(__builtin_amdgcn_tanh_f32)
  return __builtin_amdgcn_tanh_f32(x);
#else
  float ax = fabsf(x);
  float e  = __expf(-2.0f * ax);
  float r  = 1.0f - 2.0f * e * __builtin_amdgcn_rcpf(1.0f + e);
  return copysignf(r, x);
#endif
}

__device__ __forceinline__ v8f wmma_bf16(v16bf a, v16bf b, v8f c) {
  // v_wmma_f32_16x16x32_bf16: D = A(16x32) * B(32x16) + C
  return __builtin_amdgcn_wmma_f32_16x16x32_bf16(
      /*neg_a=*/false, a, /*neg_b=*/false, b,
      /*c_mod=*/(short)0, c, /*reuse_a=*/false, /*reuse_b=*/false);
}

// B-operand fragment: lane l holds column n=(l&15), K = ktile*32 + (l<16?0:16) + [0..15],
// pre-swizzled contiguously by prep kernel: 32 bytes per lane, 1KB per fragment.
__device__ __forceinline__ v16bf load_b(const unsigned short* p) {
  U32B t;
  t.lo = *(const uint4*)(p);
  t.hi = *(const uint4*)(p + 8);
  return __builtin_bit_cast(v16bf, t);
}

// A-operand fragment from LDS bf16 activations [16 rows][K], row-major.
// Lane l: row m=(l&15); K = ktile*32 + (l<16?0:16) + [0..15]  (contiguous, matches B)
__device__ __forceinline__ v16bf load_a(const unsigned short* buf, int K, int ktile, int lane) {
  int m  = lane & 15;
  int kb = ktile * 32 + ((lane < 16) ? 0 : 16);
  const unsigned short* p = buf + m * K + kb;
  U32B t;
  t.lo = *(const uint4*)(p);        // K = kb .. kb+7
  t.hi = *(const uint4*)(p + 8);    // K = kb+8 .. kb+15
  return __builtin_bit_cast(v16bf, t);
}

// ---------------------------------------------------------------------------
// Prep: convert Wf1/Wf2/Wf3 (f32, row-major [N][K], used as B = W^T) to bf16
// WMMA-B fragments in d_ws.  frag f of weight: ntile=f/ktiles, ktile=f%ktiles.
// ws layout: Wf1 frags [0,16), Wf2 [16,48), Wf3 [48,304); 512 ushort each.
// ---------------------------------------------------------------------------
__global__ void prep_weights(const float* __restrict__ Wf1,
                             const float* __restrict__ Wf2,
                             const float* __restrict__ Wf3,
                             unsigned short* __restrict__ ws) {
  int g = blockIdx.x * blockDim.x + threadIdx.x;
  if (g >= 304 * 32) return;
  int frag = g >> 5, lane = g & 31;
  const float* W; int K, f, base;
  if (frag < 16)      { W = Wf1; K = 64;  f = frag;      base = 0; }
  else if (frag < 48) { W = Wf2; K = 128; f = frag - 16; base = 16 * 512; }
  else                { W = Wf3; K = 128; f = frag - 48; base = 48 * 512; }
  int ktiles = K >> 5;
  int ntile = f / ktiles, ktile = f % ktiles;
  int n  = ntile * 16 + (lane & 15);
  int k0 = ktile * 32 + ((lane < 16) ? 0 : 16);
  unsigned short* dst = ws + base + f * 512 + lane * 16;
  #pragma unroll
  for (int i = 0; i < 16; ++i) dst[i] = f2bf(W[n * K + k0 + i]);
}

// ---------------------------------------------------------------------------
// Main kernel: 16 blocks x 256 threads (8 wave32).  Block handles 16 batch
// rows; runs the whole 63-step, 6-stage RK scan in-kernel.
// ---------------------------------------------------------------------------
__global__ void __launch_bounds__(256)
ncde_main(const float* __restrict__ ts,
          const float* __restrict__ cd, const float* __restrict__ cc,
          const float* __restrict__ cb, const float* __restrict__ ca,
          const float* __restrict__ Wi1, const float* __restrict__ bi1,
          const float* __restrict__ Wi2, const float* __restrict__ bi2,
          const float* __restrict__ bf1, const float* __restrict__ bf2,
          const float* __restrict__ bf3,
          const float* __restrict__ Wr,  const float* __restrict__ br,
          const unsigned short* __restrict__ wsu,
          float* __restrict__ out) {
  __shared__ __align__(16) float          ybuf[16 * 64];        // state y (f32)
  __shared__ __align__(16) unsigned short yjb [16 * 64];        // stage input, bf16
  __shared__ __align__(16) unsigned short h1b [16 * 128];       // layer1 act, bf16
  __shared__ __align__(16) unsigned short h2b [16 * 128];       // layer2 act, bf16
  __shared__ __align__(16) float          ksb [6 * 1024];       // RK slopes k0..k5
  __shared__ __align__(16) float          dxl [256];            // dxdt [16][16]
  __shared__ __align__(16) float          cdsh[256], ccsh[256], cbsh[256];
  __shared__ __align__(16) float          bf1s[128], bf2s[128], bf3s[1024];
  __shared__ __align__(16) float          Wrs[8 * 64];
  __shared__ float                        brs[8];

  const int tid  = threadIdx.x;
  const int wave = tid >> 5, lane = tid & 31;
  const int row0 = blockIdx.x * 16;

  const unsigned short* wf1s = wsu;             // 16 frags
  const unsigned short* wf2s = wsu + 16 * 512;  // 32 frags
  const unsigned short* wf3s = wsu + 48 * 512;  // 256 frags

  // Cache biases / readout weights in LDS.
  for (int i = tid; i < 128; i += 256) { bf1s[i] = bf1[i]; bf2s[i] = bf2[i]; }
  for (int i = tid; i < 1024; i += 256) bf3s[i] = bf3[i];
  for (int i = tid; i < 512;  i += 256) Wrs[i] = Wr[i];
  if (tid < 8) brs[tid] = br[tid];
  __syncthreads();

  // ---- y0 = softplus(x0 @ Wi1^T + bi1) @ Wi2^T + bi2 (tiny, VALU, once) ----
  float* h0 = ksb;  // scratch: 2048 f32 fits in ksb
  for (int e = tid; e < 16 * 128; e += 256) {
    int m = e >> 7, hh = e & 127;
    float acc = bi1[hh];
    const float* x0 = ca + (size_t)(row0 + m) * (NSTEP * DD);  // coeff_a[b][0][:]
    #pragma unroll
    for (int d = 0; d < 16; ++d) acc += x0[d] * Wi1[hh * 16 + d];
    h0[e] = softplus_f(acc);
  }
  __syncthreads();
  for (int e = tid; e < 16 * 64; e += 256) {
    int m = e >> 6, s = e & 63;
    float acc = bi2[s];
    for (int hh = 0; hh < 128; ++hh) acc += h0[m * 128 + hh] * Wi2[s * 128 + hh];
    ybuf[e] = acc;
  }
  __syncthreads();
  if (tid < 128) {  // out[:, t=0, :]
    int m = tid >> 3, o = tid & 7;
    float acc = brs[o];
    for (int s = 0; s < 64; ++s) acc += ybuf[m * 64 + s] * Wrs[o * 64 + s];
    out[(size_t)(row0 + m) * (TT * OO) + o] = acc;
  }
  __syncthreads();

  // RK tableau (B_SOL[6]==0 -> stage 7 never contributes: run 6 stages only)
  constexpr float CN[6] = {0.0f, 0.161f, 0.327f, 0.9f, 0.9800255409045097f, 1.0f};
  constexpr float AT[6][5] = {
    {0, 0, 0, 0, 0},
    {0.161f, 0, 0, 0, 0},
    {-0.008480655492356989f, 0.335480655492357f, 0, 0, 0},
    {2.8971530571054935f, -6.359448489975075f, 4.3622954328695815f, 0, 0},
    {5.325864828439257f, -11.748883564062828f, 7.4955393428898365f, -0.09249506636175525f, 0},
    {5.86145544294642f, -12.92096931784711f, 8.159367898576159f, -0.071584973281401f, -0.028269050394068383f}};
  constexpr float BS[6] = {0.09646076681806523f, 0.01f, 0.4798896504144996f,
                           1.379008574103742f, -3.290069515436081f, 2.324710524099774f};

  for (int t = 0; t < NSTEP; ++t) {
    float h = ts[t + 1] - ts[t];
    {  // per-step spline coefficients -> LDS (reused by all 6 stages)
      int m = tid >> 4, d = tid & 15;
      size_t idx = ((size_t)(row0 + m) * NSTEP + t) * DD + d;
      cdsh[tid] = cd[idx]; ccsh[tid] = cc[idx]; cbsh[tid] = cb[idx];
      if (t + 1 < NSTEP) {  // warm L2/L0 for next step (global_prefetch_b8)
        __builtin_prefetch(&cd[idx + DD], 0, 3);
        __builtin_prefetch(&cc[idx + DD], 0, 3);
        __builtin_prefetch(&cb[idx + DD], 0, 3);
      }
    }
    __syncthreads();

    #pragma unroll
    for (int j = 0; j < 6; ++j) {
      // ---- stage input yj (bf16) + dxdt ----
      #pragma unroll
      for (int i = 0; i < 4; ++i) {
        int e = tid + i * 256;
        float v = ybuf[e];
        #pragma unroll
        for (int mm = 0; mm < j; ++mm) v += h * AT[j][mm] * ksb[mm * 1024 + e];
        yjb[e] = f2bf(v);
      }
      {
        float frac = CN[j] * h;
        dxl[tid] = cbsh[tid] + frac * (2.0f * ccsh[tid] + 3.0f * frac * cdsh[tid]);
      }
      __syncthreads();

      // ---- GEMM1: h1 = softplus(yj @ Wf1^T + bf1), ntile == wave ----
      // C starts as inline 0 (no register init on the WMMA critical path);
      // bias is folded into the epilogue nonlinearity.
      {
        float bias = bf1s[wave * 16 + (lane & 15)];
        v8f acc = {0.f, 0.f, 0.f, 0.f, 0.f, 0.f, 0.f, 0.f};
        #pragma unroll
        for (int kt = 0; kt < 2; ++kt) {
          v16bf a = load_a(yjb, 64, kt, lane);
          v16bf b = load_b(wf1s + (wave * 2 + kt) * 512 + lane * 16);
          acc = wmma_bf16(a, b, acc);
        }
        int n = wave * 16 + (lane & 15), mb = (lane < 16) ? 0 : 8;
        #pragma unroll
        for (int r = 0; r < 8; ++r)
          h1b[(r + mb) * 128 + n] = f2bf(softplus_f(acc[r] + bias));
      }
      __syncthreads();

      // ---- GEMM2: h2 = softplus(h1 @ Wf2^T + bf2), ntile == wave ----
      {
        float bias = bf2s[wave * 16 + (lane & 15)];
        v8f acc = {0.f, 0.f, 0.f, 0.f, 0.f, 0.f, 0.f, 0.f};
        #pragma unroll
        for (int kt = 0; kt < 4; ++kt) {
          v16bf a = load_a(h1b, 128, kt, lane);
          v16bf b = load_b(wf2s + (wave * 4 + kt) * 512 + lane * 16);
          acc = wmma_bf16(a, b, acc);
        }
        int n = wave * 16 + (lane & 15), mb = (lane < 16) ? 0 : 8;
        #pragma unroll
        for (int r = 0; r < 8; ++r)
          h2b[(r + mb) * 128 + n] = f2bf(softplus_f(acc[r] + bias));
      }
      __syncthreads();

      // ---- GEMM3 + einsum: M = tanh(h2 @ Wf3^T + bf3); k = M.(S,D) dot dxdt.
      // Each 16x16 C-tile is exactly one s slice (D==16): tanh, scale by dxdt,
      // reduce over the 16-lane half-wave.
      {
        v16bf a0 = load_a(h2b, 128, 0, lane);
        v16bf a1 = load_a(h2b, 128, 1, lane);
        v16bf a2 = load_a(h2b, 128, 2, lane);
        v16bf a3 = load_a(h2b, 128, 3, lane);
        int mb = (lane < 16) ? 0 : 8, d = lane & 15;
        float dxp[8];
        #pragma unroll
        for (int r = 0; r < 8; ++r) dxp[r] = dxl[(r + mb) * 16 + d];
        #pragma unroll
        for (int it = 0; it < 8; ++it) {
          int ntile = wave * 8 + it;  // == s
          float bias = bf3s[ntile * 16 + (lane & 15)];
          v8f acc = {0.f, 0.f, 0.f, 0.f, 0.f, 0.f, 0.f, 0.f};
          const unsigned short* wb = wf3s + ntile * 4 * 512 + lane * 16;
          acc = wmma_bf16(a0, load_b(wb + 0 * 512), acc);
          acc = wmma_bf16(a1, load_b(wb + 1 * 512), acc);
          acc = wmma_bf16(a2, load_b(wb + 2 * 512), acc);
          acc = wmma_bf16(a3, load_b(wb + 3 * 512), acc);
          #pragma unroll
          for (int r = 0; r < 8; ++r) {
            float v = fast_tanh(acc[r] + bias) * dxp[r];
            v += __shfl_xor(v, 1);
            v += __shfl_xor(v, 2);
            v += __shfl_xor(v, 4);
            v += __shfl_xor(v, 8);
            if ((lane & 15) == 0) ksb[j * 1024 + (r + mb) * 64 + ntile] = v;
          }
        }
      }
      __syncthreads();
    }  // stages

    // ---- y_new = y + h * sum_j BS[j]*k_j ; emit out[:, t+1, :] ----
    #pragma unroll
    for (int i = 0; i < 4; ++i) {
      int e = tid + i * 256;
      float v = ybuf[e];
      #pragma unroll
      for (int jj = 0; jj < 6; ++jj) v += h * BS[jj] * ksb[jj * 1024 + e];
      ybuf[e] = v;
    }
    __syncthreads();
    if (tid < 128) {
      int m = tid >> 3, o = tid & 7;
      float acc = brs[o];
      for (int s = 0; s < 64; ++s) acc += ybuf[m * 64 + s] * Wrs[o * 64 + s];
      out[(size_t)(row0 + m) * (TT * OO) + (size_t)(t + 1) * OO + o] = acc;
    }
    __syncthreads();
  }
}

extern "C" void kernel_launch(void* const* d_in, const int* in_sizes, int n_in,
                              void* d_out, int out_size, void* d_ws, size_t ws_size,
                              hipStream_t stream) {
  (void)in_sizes; (void)n_in; (void)out_size; (void)ws_size;
  const float* ts  = (const float*)d_in[0];
  const float* cdp = (const float*)d_in[1];
  const float* ccp = (const float*)d_in[2];
  const float* cbp = (const float*)d_in[3];
  const float* cap = (const float*)d_in[4];
  const float* Wi1 = (const float*)d_in[5];
  const float* bi1 = (const float*)d_in[6];
  const float* Wi2 = (const float*)d_in[7];
  const float* bi2 = (const float*)d_in[8];
  const float* Wf1 = (const float*)d_in[9];
  const float* bf1 = (const float*)d_in[10];
  const float* Wf2 = (const float*)d_in[11];
  const float* bf2 = (const float*)d_in[12];
  const float* Wf3 = (const float*)d_in[13];
  const float* bf3 = (const float*)d_in[14];
  const float* Wr  = (const float*)d_in[15];
  const float* br  = (const float*)d_in[16];
  unsigned short* wsu = (unsigned short*)d_ws;   // 304 KB of bf16 weight frags
  float* out = (float*)d_out;

  prep_weights<<<(304 * 32 + 255) / 256, 256, 0, stream>>>(Wf1, Wf2, Wf3, wsu);
  ncde_main<<<16, 256, 0, stream>>>(ts, cdp, ccp, cbp, cap, Wi1, bi1, Wi2, bi2,
                                    bf1, bf2, bf3, Wr, br, wsu, out);
}
